// MultiRNNCell_62208306315770
// MI455X (gfx1250) — matrix operand order repaired and checked by reading, not compiled, not run on hardware
//
#include <hip/hip_runtime.h>
#include <hip/hip_bf16.h>

// ---------------------------------------------------------------------------
// Stacked GRU (B=128, T=128, D_IN=512, H=1024, 2 layers) for MI455X (gfx1250).
//
// Strategy:
//  * Hoist X @ Wih0^T out of the recurrence (one big 16384x512x3072 GEMM).
//  * Per timestep, the sequential work is h @ Whh0^T (N=3072) and, since
//    layer 1 feeds h as both input and hidden, h @ [Wih1;Whh1]^T (N=6144).
//  * Weights converted once to f16 and pre-packed into the exact wave32
//    WMMA B-fragment lane layout (contiguous 32B per lane per 32x16 block);
//    packed recurrent weights total 18 MB -> resident in the 192 MB L2 for
//    the whole 128-step chain.
//  * GEMM core: v_wmma_f32_16x16x32_f16, f32 accumulation, wave computes a
//    16x64 strip (A fragment reused across 4 N-tiles).
//  * Elementwise GRU gate math in f32 (sigmoid/tanh), h kept in f32 master,
//    mirrored to f16 for the next step's WMMA.
// ---------------------------------------------------------------------------

typedef __attribute__((ext_vector_type(8)))  _Float16 hv8;
typedef __attribute__((ext_vector_type(16))) _Float16 hv16;
typedef __attribute__((ext_vector_type(8)))  float    fv8;

#define B_SZ   128
#define T_SZ   128
#define D_IN   512
#define D_H    1024
#define G3     (3 * D_H)   // 3072
#define G6     (6 * D_H)   // 6144

// ---------------------------------------------------------------------------
// Pack W (N x K, row-major f32) into f16 WMMA B-fragment blocks.
// Block (nt, kc) covers B[k= kc*32 .. +31][n = nt*16 .. +15] and is stored as
// [lane 0..31][16 f16] so each lane reads its fragment as one 32-byte load.
// Lane l: n = nt*16 + (l&15); k = kc*32 + (l<16 ? 0 : 8) + (e&7) + (e>=8 ? 16 : 0), e=0..15.
// (Matches CDNA5 ISA 16-bit B layout for 16x16x32.)
// ---------------------------------------------------------------------------
__global__ void pack_b_f16(const float* __restrict__ W, _Float16* __restrict__ out,
                           int N, int K) {
    size_t idx = (size_t)blockIdx.x * blockDim.x + threadIdx.x;
    size_t tot = (size_t)N * K;
    if (idx >= tot) return;
    int e    = (int)(idx & 15);
    size_t t = idx >> 4;
    int lane = (int)(t & 31);
    t >>= 5;
    int kchunks = K >> 5;
    int kc = (int)(t % kchunks);
    int nt = (int)(t / kchunks);
    int n = nt * 16 + (lane & 15);
    int k = kc * 32 + ((lane < 16) ? 0 : 8) + (e & 7) + ((e >> 3) << 4);
    out[idx] = (_Float16)W[(size_t)n * K + k];
}

__global__ void f32_to_f16(const float* __restrict__ src, _Float16* __restrict__ dst,
                           size_t n) {
    size_t i = (size_t)blockIdx.x * blockDim.x + threadIdx.x;
    if (i < n) dst[i] = (_Float16)src[i];
}

// ---------------------------------------------------------------------------
// C[M,N] = A[M,K] (f16, row-major) x packed-W  (+ optional bias), f32 out.
// One wave -> 16(M) x 64(N) strip: 4 accumulators, A fragment reused 4x.
// M%16==0, N%64==0, K%32==0. Grid is sized so every wave is fully active
// (EXEC all-ones around WMMA, early-exit is wave-uniform).
// ---------------------------------------------------------------------------
__global__ void gemm_wmma_f16(const _Float16* __restrict__ A,
                              const _Float16* __restrict__ Bp,
                              float* __restrict__ C,
                              const float* __restrict__ bias,
                              int M, int N, int K) {
    const int wave = (int)((blockIdx.x * blockDim.x + threadIdx.x) >> 5);
    const int groups_n = N >> 6;                 // strips of 4 n-tiles
    const int total = (M >> 4) * groups_n;
    if (wave >= total) return;
    const int mt = wave / groups_n;
    const int ng = wave % groups_n;
    const int lane = threadIdx.x & 31;

    // A fragment addressing: lane l holds row m = l&15;
    // lanes 0-15: K offsets {0..7, 16..23}; lanes 16-31: {8..15, 24..31}.
    const int mloc = lane & 15;
    const int kb   = (lane < 16) ? 0 : 8;
    const _Float16* arow = A + (size_t)(mt * 16 + mloc) * K + kb;

    const int kchunks = K >> 5;
    const size_t ntile_stride = (size_t)kchunks * 512;      // f16 elems per n-tile
    const _Float16* bbase = Bp + (size_t)(ng * 4) * ntile_stride + (size_t)lane * 16;

    fv8 acc0 = {}, acc1 = {}, acc2 = {}, acc3 = {};
    for (int kc = 0; kc < kchunks; ++kc) {
        const int k0 = kc << 5;
        hv8 alo = *(const hv8*)(arow + k0);
        hv8 ahi = *(const hv8*)(arow + k0 + 16);
        hv16 a = __builtin_shufflevector(alo, ahi, 0,1,2,3,4,5,6,7,8,9,10,11,12,13,14,15);
        hv16 b0 = *(const hv16*)(bbase + 0 * ntile_stride + (size_t)kc * 512);
        hv16 b1 = *(const hv16*)(bbase + 1 * ntile_stride + (size_t)kc * 512);
        hv16 b2 = *(const hv16*)(bbase + 2 * ntile_stride + (size_t)kc * 512);
        hv16 b3 = *(const hv16*)(bbase + 3 * ntile_stride + (size_t)kc * 512);
        acc0 = __builtin_amdgcn_wmma_f32_16x16x32_f16(false, a, false, b0, (short)0, acc0, false, false);
        acc1 = __builtin_amdgcn_wmma_f32_16x16x32_f16(false, a, false, b1, (short)0, acc1, false, false);
        acc2 = __builtin_amdgcn_wmma_f32_16x16x32_f16(false, a, false, b2, (short)0, acc2, false, false);
        acc3 = __builtin_amdgcn_wmma_f32_16x16x32_f16(false, a, false, b3, (short)0, acc3, false, false);
    }

    // C layout: VGPR r, lanes 0-15 -> (M=r, N=lane); lanes 16-31 -> (M=8+r, N=lane-16)
    const int mbase = mt * 16 + ((lane >> 4) << 3);
    #pragma unroll
    for (int j = 0; j < 4; ++j) {
        const fv8 acc = (j == 0) ? acc0 : (j == 1) ? acc1 : (j == 2) ? acc2 : acc3;
        const int n = (ng * 4 + j) * 16 + (lane & 15);
        const float bv = bias ? bias[n] : 0.0f;
        float* crow = C + (size_t)mbase * N + n;
        #pragma unroll
        for (int r = 0; r < 8; ++r) crow[(size_t)r * N] = acc[r] + bv;
    }
}

// ---------------------------------------------------------------------------
// GRU gate combine (f32): r = sig(ir+hr), z = sig(iz+hz), n = tanh(in + r*hn),
// h = (1-z)*n + z*h.  gi/gh rows indexed with caller-supplied batch strides so
// the same kernel serves layer 0 (gi from precomputed GI0[t]) and layer 1
// (gi/gh are halves of the fused 6144-wide GEMM output).
// ---------------------------------------------------------------------------
__global__ void gru_combine(const float* __restrict__ gi, long long gi_bstride,
                            const float* __restrict__ gh, long long gh_bstride,
                            const float* __restrict__ bih, const float* __restrict__ bhh,
                            float* __restrict__ h, _Float16* __restrict__ h16,
                            _Float16* __restrict__ hs16, long long hs_bstride) {
    int idx = blockIdx.x * blockDim.x + threadIdx.x;   // B_SZ*D_H threads
    int b = idx >> 10;
    int j = idx & (D_H - 1);
    const float* gib = gi + (size_t)b * gi_bstride;
    const float* ghb = gh + (size_t)b * gh_bstride;
    float ir = gib[j]           + bih[j];
    float iz = gib[D_H + j]     + bih[D_H + j];
    float in = gib[2 * D_H + j] + bih[2 * D_H + j];
    float hr = ghb[j]           + bhh[j];
    float hz = ghb[D_H + j]     + bhh[D_H + j];
    float hn = ghb[2 * D_H + j] + bhh[2 * D_H + j];
    float r = 1.0f / (1.0f + expf(-(ir + hr)));
    float z = 1.0f / (1.0f + expf(-(iz + hz)));
    float n = tanhf(in + r * hn);
    float hv = (1.0f - z) * n + z * h[idx];
    h[idx]   = hv;
    h16[idx] = (_Float16)hv;
    if (hs16) hs16[(size_t)b * hs_bstride + j] = (_Float16)hv;
}

// ---------------------------------------------------------------------------
extern "C" void kernel_launch(void* const* d_in, const int* in_sizes, int n_in,
                              void* d_out, int out_size, void* d_ws, size_t ws_size,
                              hipStream_t stream) {
    (void)in_sizes; (void)n_in; (void)out_size; (void)ws_size;
    const float* X    = (const float*)d_in[0];   // (B,T,512)
    const float* Wih0 = (const float*)d_in[1];   // (3072,512)
    const float* Whh0 = (const float*)d_in[2];   // (3072,1024)
    const float* bih0 = (const float*)d_in[3];
    const float* bhh0 = (const float*)d_in[4];
    const float* Wih1 = (const float*)d_in[5];   // (1,3072,1024)
    const float* Whh1 = (const float*)d_in[6];
    const float* bih1 = (const float*)d_in[7];
    const float* bhh1 = (const float*)d_in[8];
    const float* Wout = (const float*)d_in[9];   // (512,1024)
    const float* bout = (const float*)d_in[10];
    float* out = (float*)d_out;

    // Workspace carve (256B aligned).
    char* p = (char*)d_ws;
    auto alloc = [&](size_t bytes) -> void* {
        void* r = (void*)p; p += (bytes + 255) & ~(size_t)255; return r;
    };
    _Float16* Wih0p = (_Float16*)alloc((size_t)G3 * D_IN * 2);   // 3 MB
    _Float16* Whh0p = (_Float16*)alloc((size_t)G3 * D_H * 2);    // 6 MB
    _Float16* W1p   = (_Float16*)alloc((size_t)G6 * D_H * 2);    // 12 MB  [Wih1;Whh1]
    _Float16* Woutp = (_Float16*)alloc((size_t)D_IN * D_H * 2);  // 1 MB
    _Float16* Xf16  = (_Float16*)alloc((size_t)B_SZ * T_SZ * D_IN * 2);  // 16 MB
    float*    GI0   = (float*)alloc((size_t)B_SZ * T_SZ * G3 * 4);       // 192 MB
    float*    G0    = (float*)alloc((size_t)B_SZ * G3 * 4);
    float*    G1    = (float*)alloc((size_t)B_SZ * G6 * 4);
    float*    h     = (float*)alloc((size_t)B_SZ * D_H * 4);
    _Float16* h16   = (_Float16*)alloc((size_t)B_SZ * D_H * 2);
    _Float16* hs16  = (_Float16*)alloc((size_t)B_SZ * T_SZ * D_H * 2);   // 32 MB

    auto cdiv = [](size_t a, size_t b) { return (unsigned)((a + b - 1) / b); };

    // ---- One-time packing / conversion ------------------------------------
    pack_b_f16<<<cdiv((size_t)G3 * D_IN, 256), 256, 0, stream>>>(Wih0, Wih0p, G3, D_IN);
    pack_b_f16<<<cdiv((size_t)G3 * D_H, 256), 256, 0, stream>>>(Whh0, Whh0p, G3, D_H);
    // Combined [Wih1;Whh1]: n-tiles for rows >= 3072 land contiguously after
    // the first 3072*K packed elements, so two packs produce the fused layout.
    pack_b_f16<<<cdiv((size_t)G3 * D_H, 256), 256, 0, stream>>>(Wih1, W1p, G3, D_H);
    pack_b_f16<<<cdiv((size_t)G3 * D_H, 256), 256, 0, stream>>>(Whh1, W1p + (size_t)G3 * D_H, G3, D_H);
    pack_b_f16<<<cdiv((size_t)D_IN * D_H, 256), 256, 0, stream>>>(Wout, Woutp, D_IN, D_H);
    f32_to_f16<<<cdiv((size_t)B_SZ * T_SZ * D_IN, 256), 256, 0, stream>>>(
        X, Xf16, (size_t)B_SZ * T_SZ * D_IN);

    // ---- Precompute GI0 = X @ Wih0^T for all timesteps (bias added later) --
    {
        int M = B_SZ * T_SZ, N = G3, K = D_IN;
        unsigned waves = (unsigned)((M >> 4) * (N >> 6));
        gemm_wmma_f16<<<cdiv((size_t)waves * 32, 256), 256, 0, stream>>>(
            Xf16, Wih0p, GI0, nullptr, M, N, K);
    }

    // ---- h = 0 -------------------------------------------------------------
    hipMemsetAsync(h, 0, (size_t)B_SZ * D_H * 4, stream);
    hipMemsetAsync(h16, 0, (size_t)B_SZ * D_H * 2, stream);

    // ---- Recurrence: 128 steps; recurrent weights (18 MB f16) stay in L2 ---
    const unsigned cgrid = cdiv((size_t)B_SZ * D_H, 256);
    for (int t = 0; t < T_SZ; ++t) {
        // layer 0: gh0 = h @ Whh0^T
        {
            unsigned waves = (unsigned)((B_SZ >> 4) * (G3 >> 6));   // 384
            gemm_wmma_f16<<<cdiv((size_t)waves * 32, 256), 256, 0, stream>>>(
                h16, Whh0p, G0, nullptr, B_SZ, G3, D_H);
        }
        gru_combine<<<cgrid, 256, 0, stream>>>(
            GI0 + (size_t)t * G3, (long long)T_SZ * G3,   // gi row = b*T+t
            G0, (long long)G3,
            bih0, bhh0, h, h16, nullptr, 0);

        // layer 1: [gi1|gh1] = h @ [Wih1;Whh1]^T (fused, N=6144)
        {
            unsigned waves = (unsigned)((B_SZ >> 4) * (G6 >> 6));   // 768
            gemm_wmma_f16<<<cdiv((size_t)waves * 32, 256), 256, 0, stream>>>(
                h16, W1p, G1, nullptr, B_SZ, G6, D_H);
        }
        gru_combine<<<cgrid, 256, 0, stream>>>(
            G1, (long long)G6,
            G1 + G3, (long long)G6,
            bih1, bhh1, h, h16,
            hs16 + (size_t)t * D_H, (long long)T_SZ * D_H);
    }

    // ---- out = hs @ Wout^T + bout ------------------------------------------
    {
        int M = B_SZ * T_SZ, N = D_IN, K = D_H;
        unsigned waves = (unsigned)((M >> 4) * (N >> 6));
        gemm_wmma_f16<<<cdiv((size_t)waves * 32, 256), 256, 0, stream>>>(
            hs16, Woutp, out, bout, M, N, K);
    }

    // ---- h_final appended after out ----------------------------------------
    hipMemcpyAsync(out + (size_t)B_SZ * T_SZ * D_IN, h,
                   (size_t)B_SZ * D_H * 4, hipMemcpyDeviceToDevice, stream);
}
